// RoIHeads_51625506898634
// MI455X (gfx1250) — compile-verified, matrix operand order checked by reading, not run on hardware
//
#include <hip/hip_runtime.h>
#include <math.h>

#define BB 8
#define NN 2000
#define NCLS 91
#define NFG 90
#define IMG_W_F 1333.0f
#define IMG_H_F 800.0f
#define SCORE_TH 0.05f
#define NMS_TH 0.5f
#define DETS 100
#define TOPK 2048
#define MIN_SZ 0.01f
#define XFORM_CLIP 4.135166556742356f   // log(1000/16)

#define EXP_STRIDE 97                   // odd stride -> conflict-free LDS rows

typedef __attribute__((ext_vector_type(16))) _Float16 v16h;
typedef __attribute__((ext_vector_type(8)))  float    v8f;

__device__ __forceinline__ void decode_clip(const float* __restrict__ prop,
                                            const float* __restrict__ r4,
                                            float& x1, float& y1,
                                            float& x2, float& y2) {
  float px1 = prop[0], py1 = prop[1], px2 = prop[2], py2 = prop[3];
  float w  = px2 - px1, h = py2 - py1;
  float cx = px1 + 0.5f * w, cy = py1 + 0.5f * h;
  float dx = r4[0] * 0.1f;                      // /WX
  float dy = r4[1] * 0.1f;                      // /WY
  float dw = fminf(r4[2] * 0.2f, XFORM_CLIP);   // /WW clipped
  float dh = fminf(r4[3] * 0.2f, XFORM_CLIP);   // /WH clipped
  float pcx = dx * w + cx, pcy = dy * h + cy;
  float pw = __expf(dw) * w, ph = __expf(dh) * h;
  x1 = fminf(fmaxf(pcx - 0.5f * pw, 0.f), IMG_W_F);
  y1 = fminf(fmaxf(pcy - 0.5f * ph, 0.f), IMG_H_F);
  x2 = fminf(fmaxf(pcx + 0.5f * pw, 0.f), IMG_W_F);
  y2 = fminf(fmaxf(pcy + 0.5f * ph, 0.f), IMG_H_F);
}

// ---------------------------------------------------------------------------
// K1: fused softmax (denominator via WMMA) + per-(proposal,class) decode +
//     validity mask. One wave handles 16 proposals: lane L / L+16 each supply
//     16 of the 32 K-slices of A; B == all-ones so D[m][*] = row sums exactly,
//     independent of the A K-permutation. 3 WMMAs cover C=91 (zero-padded).
//     All logit loads are unconditional (clamped index + value select) so the
//     compiler batches them under one loadcnt wait instead of exec-masked
//     single-load branches. exp(logit) values are staged in LDS and reused by
//     the scoring phase -> logits are read from global exactly once.
// ---------------------------------------------------------------------------
__global__ __launch_bounds__(256) void k_score_decode(
    const float* __restrict__ logits, const float* __restrict__ reg,
    const float* __restrict__ props, float* __restrict__ masked) {
  __shared__ float s_exp[128][EXP_STRIDE];   // exp(logit) per block proposal
  __shared__ float s_sum[128];               // softmax denominators
  const int lane  = threadIdx.x & 31;
  const int wave  = threadIdx.x >> 5;
  const int halfu = lane >> 4;
  const int wp    = wave * 16 + (lane & 15);                 // block-local row
  const int m     = (blockIdx.x * 8 + wave) * 16 + (lane & 15);
  const size_t lbase = (size_t)m * NCLS;

  v16h ones;
  #pragma unroll
  for (int e = 0; e < 16; ++e) ones[e] = (_Float16)1.0f;

  v8f acc = {};
  #pragma unroll
  for (int c = 0; c < 3; ++c) {
    v16h a;
    int kbase = 32 * c + 16 * halfu;
    #pragma unroll
    for (int e = 0; e < 16; ++e) {
      int k  = kbase + e;
      int kc = (k < NCLS) ? k : (NCLS - 1);     // clamped, unconditional load
      float x = __expf(logits[lbase + kc]);
      x = (k < NCLS) ? x : 0.0f;                // value select, no branch
      s_exp[wp][k] = x;                         // k <= 95 < EXP_STRIDE
      a[e] = (_Float16)x;
    }
    acc = __builtin_amdgcn_wmma_f32_16x16x32_f16(
        false, a, false, ones, (short)0, acc, false, false);
  }
  // D layout: acc[r] on lane 0 = rowsum(r), on lane 16 = rowsum(8+r)
  if ((lane & 15) == 0) {
    #pragma unroll
    for (int r = 0; r < 8; ++r)
      s_sum[wave * 16 + halfu * 8 + r] = acc[r];
  }
  __syncthreads();
  const float inv = 1.0f / s_sum[wp];

  // score + decode + mask: two lanes per proposal, 45 fg classes each
  const int b = m / NN, n = m % NN;
  const float* prop = props + (size_t)m * 4;
  float pr[4] = {prop[0], prop[1], prop[2], prop[3]};
  float* mrow = masked + (size_t)b * (NN * NFG) + (size_t)n * NFG;
  const float* rrow = reg + (size_t)m * (NCLS * 4);
  const int c0 = 1 + halfu * 45;
  for (int cc = 0; cc < 45; ++cc) {
    int c = c0 + cc;
    float s = s_exp[wp][c] * inv;               // reuse staged exp from LDS
    float x1, y1, x2, y2;
    decode_clip(pr, rrow + c * 4, x1, y1, x2, y2);
    float wsz = x2 - x1, hsz = y2 - y1;
    bool valid = (s > SCORE_TH) && (wsz >= MIN_SZ) && (hsz >= MIN_SZ);
    mrow[c - 1] = valid ? s : -1.0f;
  }
}

// ---------------------------------------------------------------------------
// K2: per-image top-2048 via 4096-bin float-bit histogram threshold,
//     LDS compaction, then 2048-wide bitonic sort (descending). 1 block/image.
// ---------------------------------------------------------------------------
__global__ __launch_bounds__(1024) void k_topk(
    const float* __restrict__ masked, float* __restrict__ cand_score,
    int* __restrict__ cand_idx) {
  __shared__ int   hist[4096];
  __shared__ float cs[TOPK];
  __shared__ int   ci[TOPK];
  __shared__ int   sh_T, sh_cnt;
  const int b = blockIdx.x, tid = threadIdx.x;
  const float* ms = masked + (size_t)b * (NN * NFG);
  const int TOT = NN * NFG;

  for (int i = tid; i < 4096; i += 1024) hist[i] = 0;
  __syncthreads();
  for (int i = tid; i < TOT; i += 1024) {
    float s = ms[i];
    if (s > 0.f)
      atomicAdd(&hist[(__float_as_uint(s) >> 19) & 0xFFF], 1);
  }
  __syncthreads();
  if (tid == 0) {
    int acc = 0, T = -1;
    for (int bin = 4095; bin >= 0; --bin) {
      acc += hist[bin];
      if (acc >= TOPK) { T = bin; break; }
    }
    sh_T = T; sh_cnt = 0;
  }
  __syncthreads();
  const int T = sh_T;
  for (int i = tid; i < TOT; i += 1024) {       // strictly-above threshold
    float s = ms[i];
    if (s > 0.f) {
      int bin = (__float_as_uint(s) >> 19) & 0xFFF;
      if (T < 0 || bin > T) {
        int pos = atomicAdd(&sh_cnt, 1);
        if (pos < TOPK) { cs[pos] = s; ci[pos] = i; }
      }
    }
  }
  __syncthreads();
  if (T >= 0) {                                  // fill from threshold bin
    for (int i = tid; i < TOT; i += 1024) {
      float s = ms[i];
      if (s > 0.f && ((__float_as_uint(s) >> 19) & 0xFFF) == (unsigned)T) {
        int pos = atomicAdd(&sh_cnt, 1);
        if (pos < TOPK) { cs[pos] = s; ci[pos] = i; }
      }
    }
  }
  __syncthreads();
  int total = sh_cnt; if (total > TOPK) total = TOPK;
  for (int i = tid; i < TOPK; i += 1024)
    if (i >= total) { cs[i] = -1.0f; ci[i] = 0; }
  // bitonic sort, descending by score
  for (int k = 2; k <= TOPK; k <<= 1) {
    for (int j = k >> 1; j > 0; j >>= 1) {
      __syncthreads();
      for (int t = tid; t < TOPK; t += 1024) {
        int ixj = t ^ j;
        if (ixj > t) {
          bool descSeg = ((t & k) == 0);
          float a = cs[t], bsc = cs[ixj];
          bool sw = descSeg ? (a < bsc) : (a > bsc);
          if (sw) {
            cs[t] = bsc; cs[ixj] = a;
            int ti = ci[t]; ci[t] = ci[ixj]; ci[ixj] = ti;
          }
        }
      }
    }
  }
  __syncthreads();
  for (int i = tid; i < TOPK; i += 1024) {
    cand_score[(size_t)b * TOPK + i] = cs[i];
    cand_idx  [(size_t)b * TOPK + i] = ci[i];
  }
}

// ---------------------------------------------------------------------------
// K3: decode only the surviving 8x2048 candidates (re-derive box from
//     proposal + class regression; cheaper than storing 180k boxes/image).
// ---------------------------------------------------------------------------
__global__ __launch_bounds__(256) void k_decode_cand(
    const float* __restrict__ reg, const float* __restrict__ props,
    const float* __restrict__ cand_score, const int* __restrict__ cand_idx,
    float* __restrict__ cand_box, int* __restrict__ cand_label) {
  int g = blockIdx.x * 256 + threadIdx.x;
  if (g >= BB * TOPK) return;
  int b = g / TOPK;
  float s = cand_score[g];
  float x1 = 0.f, y1 = 0.f, x2 = 0.f, y2 = 0.f;
  int lab = 0;
  if (s > 0.f) {
    int idx = cand_idx[g];
    int n = idx / NFG, c = idx % NFG + 1;
    int m = b * NN + n;
    float pr[4] = {props[(size_t)m*4+0], props[(size_t)m*4+1],
                   props[(size_t)m*4+2], props[(size_t)m*4+3]};
    decode_clip(pr, reg + (size_t)m * (NCLS * 4) + (size_t)c * 4,
                x1, y1, x2, y2);
    lab = c;
  }
  float* ob = cand_box + (size_t)g * 4;
  ob[0] = x1; ob[1] = y1; ob[2] = x2; ob[3] = y2;
  cand_label[g] = lab;
}

// ---------------------------------------------------------------------------
// K4: class-wise greedy NMS + emit top-100. Boxes staged to LDS with async
//     global->LDS b128 loads (ASYNCcnt), labels/keep via normal loads.
//     Per-class coordinate offsets reduce to a label-equality test since the
//     reference offset fully separates classes (gap > max coordinate).
// ---------------------------------------------------------------------------
__global__ __launch_bounds__(1024) void k_nms_out(
    const float* __restrict__ cand_score, const float* __restrict__ cand_box,
    const int* __restrict__ cand_label, float* __restrict__ out) {
  __shared__ float bbox[TOPK * 4];
  __shared__ float area[TOPK];
  __shared__ int   lab[TOPK];
  __shared__ int   keep[TOPK];
  const int b = blockIdx.x, tid = threadIdx.x;

  // async-stage 2048 boxes (16B each) into LDS
  for (int t = tid; t < TOPK; t += 1024) {
    unsigned lds = (unsigned)(unsigned long long)(void*)(&bbox[t * 4]);
    const float* gp = cand_box + ((size_t)b * TOPK + t) * 4;
    asm volatile("global_load_async_to_lds_b128 %0, %1, off"
                 :: "v"(lds), "v"(gp) : "memory");
  }
  for (int t = tid; t < TOPK; t += 1024) {
    lab[t]  = cand_label[(size_t)b * TOPK + t];
    keep[t] = (cand_score[(size_t)b * TOPK + t] > 0.f) ? 1 : 0;
  }
#if __has_builtin(__builtin_amdgcn_s_wait_asynccnt)
  __builtin_amdgcn_s_wait_asynccnt(0);
#else
  asm volatile("s_wait_asynccnt 0" ::: "memory");
#endif
  __syncthreads();
  for (int t = tid; t < TOPK; t += 1024)
    area[t] = (bbox[t*4+2] - bbox[t*4+0]) * (bbox[t*4+3] - bbox[t*4+1]);
  __syncthreads();

  // greedy suppression in score order (candidates already sorted desc)
  for (int i = 0; i < TOPK; ++i) {
    if (keep[i]) {                              // uniform across block
      float ix1 = bbox[i*4+0], iy1 = bbox[i*4+1];
      float ix2 = bbox[i*4+2], iy2 = bbox[i*4+3];
      float ia = area[i]; int il = lab[i];
      for (int t = tid; t < TOPK; t += 1024) {
        if (t > i && keep[t] && lab[t] == il) {
          float xx1 = fmaxf(ix1, bbox[t*4+0]);
          float yy1 = fmaxf(iy1, bbox[t*4+1]);
          float xx2 = fminf(ix2, bbox[t*4+2]);
          float yy2 = fminf(iy2, bbox[t*4+3]);
          float iw = fmaxf(0.f, xx2 - xx1), ih = fmaxf(0.f, yy2 - yy1);
          float inter = iw * ih;
          float iou = inter / (ia + area[t] - inter);
          if (iou > NMS_TH) keep[t] = 0;
        }
      }
    }
    __syncthreads();
  }

  // emit first 100 kept (sorted order == top-100 of kept scores)
  if (tid == 0) {
    float* ob = out;                                   // [B,100,4]
    float* os = out + (size_t)BB * DETS * 4;           // [B,100]
    float* ol = os + (size_t)BB * DETS;                // [B,100]
    int cnt = 0;
    for (int i = 0; i < TOPK && cnt < DETS; ++i) {
      if (keep[i]) {
        ob[(size_t)b*DETS*4 + cnt*4 + 0] = bbox[i*4+0];
        ob[(size_t)b*DETS*4 + cnt*4 + 1] = bbox[i*4+1];
        ob[(size_t)b*DETS*4 + cnt*4 + 2] = bbox[i*4+2];
        ob[(size_t)b*DETS*4 + cnt*4 + 3] = bbox[i*4+3];
        os[(size_t)b*DETS + cnt] = cand_score[(size_t)b*TOPK + i];
        ol[(size_t)b*DETS + cnt] = (float)lab[i];
        ++cnt;
      }
    }
    for (; cnt < DETS; ++cnt) {
      ob[(size_t)b*DETS*4 + cnt*4 + 0] = 0.f;
      ob[(size_t)b*DETS*4 + cnt*4 + 1] = 0.f;
      ob[(size_t)b*DETS*4 + cnt*4 + 2] = 0.f;
      ob[(size_t)b*DETS*4 + cnt*4 + 3] = 0.f;
      os[(size_t)b*DETS + cnt] = -1.0f;
      ol[(size_t)b*DETS + cnt] = 0.f;
    }
  }
}

extern "C" void kernel_launch(void* const* d_in, const int* in_sizes, int n_in,
                              void* d_out, int out_size, void* d_ws, size_t ws_size,
                              hipStream_t stream) {
  (void)in_sizes; (void)n_in; (void)out_size; (void)ws_size;
  const float* logits = (const float*)d_in[0];   // [B*N, 91]
  const float* reg    = (const float*)d_in[1];   // [B*N, 364]
  const float* props  = (const float*)d_in[2];   // [B, N, 4]
  float* out = (float*)d_out;                    // 3200 + 800 + 800 floats

  char* ws = (char*)d_ws;
  size_t off = 0;
  float* masked = (float*)(ws + off); off += (size_t)BB * NN * NFG * 4;   // 5.76 MB
  float* cscore = (float*)(ws + off); off += (size_t)BB * TOPK * 4;
  int*   cidx   = (int*)  (ws + off); off += (size_t)BB * TOPK * 4;
  float* cbox   = (float*)(ws + off); off += (size_t)BB * TOPK * 16;
  int*   clab   = (int*)  (ws + off); off += (size_t)BB * TOPK * 4;

  k_score_decode<<<(BB * NN) / 128, 256, 0, stream>>>(logits, reg, props, masked);
  k_topk<<<BB, 1024, 0, stream>>>(masked, cscore, cidx);
  k_decode_cand<<<(BB * TOPK + 255) / 256, 256, 0, stream>>>(
      reg, props, cscore, cidx, cbox, clab);
  k_nms_out<<<BB, 1024, 0, stream>>>(cscore, cbox, clab, out);
}
